// AttentionDecoderModel_63058709840634
// MI455X (gfx1250) — matrix operand order repaired and checked by reading, not compiled, not run on hardware
//
#include <hip/hip_runtime.h>
#include <math.h>

// ---------------------------------------------------------------------------
// Model dims (compile-time constants from the reference)
// ---------------------------------------------------------------------------
#define BB   8
#define TT   512
#define SS   1024
#define DD   512
#define ADIM 512
#define HH   8
#define HD   64
#define FFD  2048
#define LL   6
#define VV   8000

// ---------------------------------------------------------------------------
// Feature probes (evaluated per compilation pass; device pass sees gfx1250)
// ---------------------------------------------------------------------------
#ifndef __has_builtin
#define __has_builtin(x) 0
#endif
#if __has_builtin(__builtin_amdgcn_global_load_async_to_lds_b128)
#define HAVE_ASYNC_LDS 1
#else
#define HAVE_ASYNC_LDS 0
#endif
#if __has_builtin(__builtin_amdgcn_cvt_pk_bf16_f32)
#define HAVE_PK_BF16 1
#else
#define HAVE_PK_BF16 0
#endif

// ---------------------------------------------------------------------------
// WMMA types / helpers (CDNA5 gfx1250, wave32)
// ---------------------------------------------------------------------------
typedef unsigned short u16;
typedef __attribute__((ext_vector_type(16))) __bf16  v16bf;
typedef __attribute__((ext_vector_type(8)))  float   v8f;
typedef __attribute__((ext_vector_type(8)))  u16     us8;
typedef __attribute__((ext_vector_type(16))) u16     us16;
typedef __attribute__((ext_vector_type(4)))  int     v4i;

union FragU {
    us8  h[2];
    us16 v;
};

__device__ __forceinline__ u16 f2bf(float f) {
    unsigned int u = __builtin_bit_cast(unsigned int, f);
    unsigned int r = u + 0x7FFFu + ((u >> 16) & 1u);   // RTNE
    return (u16)(r >> 16);
}

__device__ __forceinline__ unsigned int pk2bf(float a, float b) {
#if HAVE_PK_BF16
    auto r = __builtin_amdgcn_cvt_pk_bf16_f32(a, b);
    return __builtin_bit_cast(unsigned int, r);
#else
    return (unsigned int)f2bf(a) | ((unsigned int)f2bf(b) << 16);
#endif
}

// Build a 16-element bf16 fragment from two 16-byte LDS chunks.
__device__ __forceinline__ v16bf make_frag(const u16* p0, const u16* p1) {
    FragU u;
    u.h[0] = *(const us8*)p0;
    u.h[1] = *(const us8*)p1;
    return __builtin_bit_cast(v16bf, u.v);
}

__device__ __forceinline__ v8f wmma_bf16(v16bf a, v16bf b, v8f c) {
    return __builtin_amdgcn_wmma_f32_16x16x32_bf16(
        false, a, false, b, (short)0, c, false, false);
}

#define V8F_ZERO {0.f,0.f,0.f,0.f,0.f,0.f,0.f,0.f}
#define US8_ZERO {0,0,0,0,0,0,0,0}

#if HAVE_ASYNC_LDS
// Builtin's pointer params are int4-typed (revealed by the round-2 probe):
// (global int4*, lds int4*, imm offset, imm cpol)
typedef __attribute__((address_space(1))) v4i gv4i;
typedef __attribute__((address_space(3))) v4i lv4i;
__device__ __forceinline__ void async_b128(const u16* g, u16* lds) {
    __builtin_amdgcn_global_load_async_to_lds_b128((gv4i*)g, (lv4i*)lds, 0, 0);
}
#if __has_builtin(__builtin_amdgcn_s_wait_asynccnt)
#define ASYNC_WAIT() __builtin_amdgcn_s_wait_asynccnt(0)
#else
#define ASYNC_WAIT() asm volatile("s_wait_asynccnt 0x0" ::: "memory")
#endif
#endif

// Stage 16 bytes (8 bf16) global -> LDS.
__device__ __forceinline__ void stage8(const u16* g, u16* lds) {
#if HAVE_ASYNC_LDS
    async_b128(g, lds);
#else
    *(us8*)lds = *(const us8*)g;
#endif
}

// ---------------------------------------------------------------------------
// f32 -> bf16 bulk conversion (pairs)
// ---------------------------------------------------------------------------
__global__ __launch_bounds__(256)
void cvt_f32_bf16_kernel(const float* __restrict__ src, u16* __restrict__ dst,
                         int n2) {
    const int i = blockIdx.x * 256 + threadIdx.x;
    if (i < n2) {
        const float2 f = ((const float2*)src)[i];
        ((unsigned int*)dst)[i] = pk2bf(f.x, f.y);
    }
}

// ---------------------------------------------------------------------------
// Embedding + sinusoidal positional encoding (f32 residual stream)
// ---------------------------------------------------------------------------
__global__ __launch_bounds__(256)
void embed_pe_kernel(const int* __restrict__ x, const float* __restrict__ embed,
                     float* __restrict__ hout) {
    const int r = blockIdx.x;           // 0 .. B*T-1
    const int b = r >> 9;               // T = 512
    const int t = r & 511;
    const int tok = x[b * TT + t];
    const float* e = embed + (size_t)tok * DD;
    const float sq = 22.62741699796952f;              // sqrt(512)
    const float c  = -9.210340371976184f / (float)DD; // -ln(10000)/D
#pragma unroll
    for (int j = 0; j < 2; ++j) {
        const int d = threadIdx.x + j * 256;
        const float div = __expf((float)((d >> 1) * 2) * c);
        const float ang = (float)t * div;
        const float pe  = (d & 1) ? __cosf(ang) : __sinf(ang);
        hout[(size_t)r * DD + d] = e[d] * sq + pe;
    }
}

// ---------------------------------------------------------------------------
// LayerNorm over 512 features -> bf16 output, one row per block (256 threads)
// ---------------------------------------------------------------------------
__global__ __launch_bounds__(256)
void layernorm_kernel(const float* __restrict__ x, const float* __restrict__ w,
                      const float* __restrict__ bsh, u16* __restrict__ out) {
    __shared__ float red[256];
    const int tid = threadIdx.x;
    const int row = blockIdx.x;
    const float* xr = x + (size_t)row * DD;
    const float v0 = xr[tid];
    const float v1 = xr[tid + 256];

    red[tid] = v0 + v1;
    __syncthreads();
    for (int o = 128; o > 0; o >>= 1) {
        if (tid < o) red[tid] += red[tid + o];
        __syncthreads();
    }
    const float mean = red[0] * (1.0f / (float)DD);
    __syncthreads();

    const float d0 = v0 - mean, d1 = v1 - mean;
    red[tid] = d0 * d0 + d1 * d1;
    __syncthreads();
    for (int o = 128; o > 0; o >>= 1) {
        if (tid < o) red[tid] += red[tid + o];
        __syncthreads();
    }
    const float rs = rsqrtf(red[0] * (1.0f / (float)DD) + 1e-5f);

    out[(size_t)row * DD + tid]       = f2bf(d0 * rs * w[tid]       + bsh[tid]);
    out[(size_t)row * DD + tid + 256] = f2bf(d1 * rs * w[tid + 256] + bsh[tid + 256]);
}

// ---------------------------------------------------------------------------
// WMMA GEMM:  C[M,N] = A[M,K] * W[N,K]^T + bias
//   A, W are bf16; output either bf16 (Cb) or f32 (+residual) (Cf).
//   block = 256 threads (8 waves), tile 128x128, BK = 32.
//   Tile staging via async-to-LDS (ASYNCcnt) when available.
// ---------------------------------------------------------------------------
__global__ __launch_bounds__(256)
void gemm_bias_kernel(const u16* __restrict__ A, const u16* __restrict__ W,
                      const float* __restrict__ bias, const float* res,
                      float* Cf, u16* Cb, int M, int N, int K, int act) {
    __shared__ __align__(16) u16 sA[128][32];
    __shared__ __align__(16) u16 sW[128][32];

    const int tid  = threadIdx.x;
    const int lane = tid & 31;
    const int wave = tid >> 5;
    const int wm   = wave >> 2;         // 0..1  (64 rows each)
    const int wn   = wave & 3;          // 0..3  (32 cols each)
    const int tileM = blockIdx.y * 128;
    const int tileN = blockIdx.x * 128;

    v8f acc[4][2];
#pragma unroll
    for (int i = 0; i < 4; ++i)
#pragma unroll
        for (int j = 0; j < 2; ++j) { v8f z = V8F_ZERO; acc[i][j] = z; }

    const int lr = tid >> 1;            // 0..127 (tile row)
    const int lc = (tid & 1) * 16;      // 0 or 16 (k within BK)
    const int mrow = lane & 15;
    const int lh   = lane >> 4;
    const bool okw = (tileN + lr) < N;  // M is always a multiple of 128 here

    for (int k0 = 0; k0 < K; k0 += 32) {
        // ---- stage A tile (pure bf16 copy) ----
        {
            const u16* src = A + (size_t)(tileM + lr) * K + k0 + lc;
            stage8(src,     &sA[lr][lc]);
            stage8(src + 8, &sA[lr][lc + 8]);
        }
        // ---- stage W tile (guard N edge for the vocab projection) ----
        if (okw) {
            const u16* src = W + (size_t)(tileN + lr) * K + k0 + lc;
            stage8(src,     &sW[lr][lc]);
            stage8(src + 8, &sW[lr][lc + 8]);
        } else {
            us8 z = US8_ZERO;
            *(us8*)&sW[lr][lc]     = z;
            *(us8*)&sW[lr][lc + 8] = z;
        }
#if HAVE_ASYNC_LDS
        ASYNC_WAIT();
#endif
        __syncthreads();

        v16bf afr[4];
#pragma unroll
        for (int i = 0; i < 4; ++i) {
            const u16* base = &sA[wm * 64 + i * 16 + mrow][lh * 8];
            afr[i] = make_frag(base, base + 16);
        }
        v16bf bfr[2];
#pragma unroll
        for (int j = 0; j < 2; ++j) {
            const u16* base = &sW[wn * 32 + j * 16 + mrow][lh * 8];
            bfr[j] = make_frag(base, base + 16);
        }
#pragma unroll
        for (int i = 0; i < 4; ++i)
#pragma unroll
            for (int j = 0; j < 2; ++j)
                acc[i][j] = wmma_bf16(afr[i], bfr[j], acc[i][j]);
        __syncthreads();
    }

    // ---- epilogue: bias, optional Swish, optional residual, f32/bf16 out ----
#pragma unroll
    for (int i = 0; i < 4; ++i)
#pragma unroll
        for (int j = 0; j < 2; ++j)
#pragma unroll
            for (int e = 0; e < 8; ++e) {
                const int m = tileM + wm * 64 + i * 16 + ((lane < 16) ? e : e + 8);
                const int n = tileN + wn * 32 + j * 16 + (lane & 15);
                if (m < M && n < N) {
                    float v = acc[i][j][e] + bias[n];
                    if (act == 1) v = v / (1.f + __expf(-v));   // Swish
                    if (Cf) {
                        if (res) v += res[(size_t)m * N + n];
                        Cf[(size_t)m * N + n] = v;
                    } else {
                        Cb[(size_t)m * N + n] = f2bf(v);
                    }
                }
            }
}

// ---------------------------------------------------------------------------
// Flash-style WMMA attention (bf16 Q/K/V in, bf16 context out).
//   block = 128 threads (4 waves); grid = (Tq/64, B*H)
//   wave w owns 16 queries; loops over 64-key blocks with online softmax.
// ---------------------------------------------------------------------------
__global__ __launch_bounds__(128)
void attn_kernel(const u16* __restrict__ Q, const u16* __restrict__ Kp,
                 const u16* __restrict__ Vp, u16* __restrict__ O,
                 const int* __restrict__ lens, int Tq, int Sk, int causal,
                 float scale) {
    __shared__ __align__(16) u16 sQ[64][64];
    __shared__ __align__(16) u16 sK[64][64];   // [key][dim]
    __shared__ __align__(16) u16 sVT[64][64];  // [dim][key]
    __shared__ __align__(16) u16 sP[4][16][64];

    const int tid  = threadIdx.x;
    const int lane = tid & 31;
    const int wave = tid >> 5;
    const int b  = blockIdx.y >> 3;     // H = 8
    const int h  = blockIdx.y & 7;
    const int q0 = blockIdx.x * 64;
    const int len = lens[b];

    // ---- stage Q tile (pure bf16 copy, 64 queries x 64 dims) ----
    {
        const int qr = tid >> 1;
        const int ch = (tid & 1) * 32;
        const u16* src = Q + ((size_t)(b * Tq + q0 + qr)) * ADIM + h * HD + ch;
#pragma unroll
        for (int c = 0; c < 32; c += 8) stage8(src + c, &sQ[qr][ch + c]);
    }
#if HAVE_ASYNC_LDS
    ASYNC_WAIT();
#endif
    __syncthreads();

    const int mrow = lane & 15;
    const int lh   = lane >> 4;

    v16bf aQ[2];
#pragma unroll
    for (int ks = 0; ks < 2; ++ks) {
        const u16* base = &sQ[wave * 16 + mrow][ks * 32 + lh * 8];
        aQ[ks] = make_frag(base, base + 16);
    }

    float mrun[8], lrun[8];
#pragma unroll
    for (int e = 0; e < 8; ++e) { mrun[e] = -1e30f; lrun[e] = 0.f; }
    v8f oacc[4];
#pragma unroll
    for (int nf = 0; nf < 4; ++nf) { v8f z = V8F_ZERO; oacc[nf] = z; }

    for (int s0 = 0; s0 < Sk; s0 += 64) {
        __syncthreads();
        // ---- stage K block [key][dim] (async) + V transposed [dim][key] ----
        {
            const int kr = tid >> 1;
            const int ch = (tid & 1) * 32;
            const u16* srcK = Kp + ((size_t)(b * Sk + s0 + kr)) * ADIM + h * HD + ch;
            const u16* srcV = Vp + ((size_t)(b * Sk + s0 + kr)) * ADIM + h * HD + ch;
#pragma unroll
            for (int c = 0; c < 32; c += 8) {
                stage8(srcK + c, &sK[kr][ch + c]);
                const us8 v = *(const us8*)(srcV + c);
#pragma unroll
                for (int j = 0; j < 8; ++j) sVT[ch + c + j][kr] = v[j];
            }
        }
#if HAVE_ASYNC_LDS
        ASYNC_WAIT();
#endif
        __syncthreads();

        // ---- S = Q * K^T  (16 queries x 64 keys per wave) ----
        v8f sacc[4];
#pragma unroll
        for (int nf = 0; nf < 4; ++nf) { v8f z = V8F_ZERO; sacc[nf] = z; }
#pragma unroll
        for (int ks = 0; ks < 2; ++ks)
#pragma unroll
            for (int nf = 0; nf < 4; ++nf) {
                const u16* base = &sK[nf * 16 + mrow][ks * 32 + lh * 8];
                v16bf bf = make_frag(base, base + 16);
                sacc[nf] = wmma_bf16(aQ[ks], bf, sacc[nf]);
            }

        // ---- mask + online softmax (rows live in 16-lane halves) ----
#pragma unroll
        for (int e = 0; e < 8; ++e) {
            const int r16  = (lane < 16) ? e : e + 8;
            const int trow = q0 + wave * 16 + r16;
            float sv[4];
            float rm = -1e30f;
#pragma unroll
            for (int nf = 0; nf < 4; ++nf) {
                const int scol = s0 + nf * 16 + (lane & 15);
                const bool masked = (scol >= len) || (causal && (scol > trow));
                sv[nf] = masked ? -1e30f : sacc[nf][e] * scale;
                rm = fmaxf(rm, sv[nf]);
            }
            rm = fmaxf(rm, __shfl_xor(rm, 1));
            rm = fmaxf(rm, __shfl_xor(rm, 2));
            rm = fmaxf(rm, __shfl_xor(rm, 4));
            rm = fmaxf(rm, __shfl_xor(rm, 8));
            const float mnew = fmaxf(mrun[e], rm);
            const float corr = __expf(mrun[e] - mnew);
            mrun[e] = mnew;
            float psum = 0.f;
#pragma unroll
            for (int nf = 0; nf < 4; ++nf) {
                const float p = __expf(sv[nf] - mnew);
                psum += p;
                sP[wave][r16][nf * 16 + (lane & 15)] = f2bf(p);
            }
            psum += __shfl_xor(psum, 1);
            psum += __shfl_xor(psum, 2);
            psum += __shfl_xor(psum, 4);
            psum += __shfl_xor(psum, 8);
            lrun[e] = lrun[e] * corr + psum;
#pragma unroll
            for (int nf = 0; nf < 4; ++nf) oacc[nf][e] *= corr;
        }

        // ---- O += P * V  (per-wave sP -> A-frags; DS in-order per wave) ----
#pragma unroll
        for (int ks = 0; ks < 2; ++ks) {
            const u16* pb = &sP[wave][mrow][ks * 32 + lh * 8];
            v16bf aP = make_frag(pb, pb + 16);
#pragma unroll
            for (int nf = 0; nf < 4; ++nf) {
                const u16* vb = &sVT[nf * 16 + mrow][ks * 32 + lh * 8];
                v16bf bV = make_frag(vb, vb + 16);
                oacc[nf] = wmma_bf16(aP, bV, oacc[nf]);
            }
        }
    }

    // ---- normalize + write bf16 context [B*Tq, AD] ----
#pragma unroll
    for (int nf = 0; nf < 4; ++nf)
#pragma unroll
        for (int e = 0; e < 8; ++e) {
            const int r16  = (lane < 16) ? e : e + 8;
            const int trow = q0 + wave * 16 + r16;
            const int d    = h * HD + nf * 16 + (lane & 15);
            O[((size_t)(b * Tq + trow)) * ADIM + d] = f2bf(oacc[nf][e] / lrun[e]);
        }
}

// ---------------------------------------------------------------------------
// Host orchestration
// ---------------------------------------------------------------------------
enum {
    LN1W, LN1B, SAQW, SAQB, SAKW, SAKB, SAVW, SAVB, SAOW, SAOB,
    LN2W, LN2B, CAQW, CAQB, CAKW, CAKB, CAVW, CAVB, CAOW, CAOB,
    LN3W, LN3B, FF1W, FF1B, FF2W, FF2B, NPARAM
};

extern "C" void kernel_launch(void* const* d_in, const int* in_sizes, int n_in,
                              void* d_out, int out_size, void* d_ws, size_t ws_size,
                              hipStream_t stream) {
    (void)in_sizes; (void)n_in; (void)out_size; (void)ws_size;

    const int*   x        = (const int*)d_in[0];
    const int*   x_lens   = (const int*)d_in[1];
    const float* memory   = (const float*)d_in[2];
    const int*   mem_lens = (const int*)d_in[3];
    const float* embed    = (const float*)d_in[4];
    const float* P[NPARAM];
    for (int i = 0; i < NPARAM; ++i) P[i] = (const float*)d_in[5 + i];
    const float* out_w = (const float*)d_in[5 + NPARAM];
    const float* out_b = (const float*)d_in[6 + NPARAM];

    const int RQ = BB * TT;     // 4096 decoder rows
    const int RM = BB * SS;     // 8192 memory rows

    // ---- workspace layout ----
    float* h = (float*)d_ws;                         // f32 residual stream
    u16* u = (u16*)(h + (size_t)RQ * DD);
    u16* lnb = u;  u += (size_t)RQ * DD;
    u16* qb  = u;  u += (size_t)RQ * DD;
    u16* kb  = u;  u += (size_t)RM * DD;
    u16* vb  = u;  u += (size_t)RM * DD;
    u16* ctx = u;  u += (size_t)RQ * DD;
    u16* ffb = u;  u += (size_t)RQ * FFD;
    u16* mbf = u;  u += (size_t)RM * DD;
    u16* hbf = u;  u += (size_t)RQ * DD;
    // bf16 weight pool
    u16* wSAQ = u; u += (size_t)LL * DD * DD;
    u16* wSAK = u; u += (size_t)LL * DD * DD;
    u16* wSAV = u; u += (size_t)LL * DD * DD;
    u16* wSAO = u; u += (size_t)LL * DD * DD;
    u16* wCAQ = u; u += (size_t)LL * DD * DD;
    u16* wCAK = u; u += (size_t)LL * DD * DD;
    u16* wCAV = u; u += (size_t)LL * DD * DD;
    u16* wCAO = u; u += (size_t)LL * DD * DD;
    u16* wFF1 = u; u += (size_t)LL * FFD * DD;
    u16* wFF2 = u; u += (size_t)LL * DD * FFD;
    u16* wOUT = u; u += (size_t)VV * DD;

    auto cvt = [&](const float* s, u16* d, size_t n) {
        const int n2 = (int)(n / 2);
        cvt_f32_bf16_kernel<<<(n2 + 255) / 256, 256, 0, stream>>>(s, d, n2);
    };

    // ---- one-time conversions (weights + encoder memory) ----
    cvt(P[SAQW], wSAQ, (size_t)LL * DD * DD);
    cvt(P[SAKW], wSAK, (size_t)LL * DD * DD);
    cvt(P[SAVW], wSAV, (size_t)LL * DD * DD);
    cvt(P[SAOW], wSAO, (size_t)LL * DD * DD);
    cvt(P[CAQW], wCAQ, (size_t)LL * DD * DD);
    cvt(P[CAKW], wCAK, (size_t)LL * DD * DD);
    cvt(P[CAVW], wCAV, (size_t)LL * DD * DD);
    cvt(P[CAOW], wCAO, (size_t)LL * DD * DD);
    cvt(P[FF1W], wFF1, (size_t)LL * FFD * DD);
    cvt(P[FF2W], wFF2, (size_t)LL * DD * FFD);
    cvt(out_w,   wOUT, (size_t)VV * DD);
    cvt(memory,  mbf,  (size_t)RM * DD);

    const float scale = 0.125f;              // 1/sqrt(HD)
    embed_pe_kernel<<<RQ, 256, 0, stream>>>(x, embed, h);

    const dim3 gD(DD / 128, RQ / 128);       // (4, 32)
    const dim3 gDM(DD / 128, RM / 128);      // (4, 64)
    const dim3 gFF1(FFD / 128, RQ / 128);    // (16, 32)
    const dim3 gAttn(TT / 64, BB * HH);      // (8, 64)

    for (int l = 0; l < LL; ++l) {
        const size_t oD  = (size_t)l * DD;
        const size_t oDD = (size_t)l * DD * DD;
        const size_t oF  = (size_t)l * FFD;
        const size_t oFD = (size_t)l * FFD * DD;

        // --- self attention ---
        layernorm_kernel<<<RQ, 256, 0, stream>>>(h, P[LN1W] + oD, P[LN1B] + oD, lnb);
        gemm_bias_kernel<<<gD, 256, 0, stream>>>(lnb, wSAQ + oDD, P[SAQB] + oD,
                                                 nullptr, nullptr, qb, RQ, ADIM, DD, 0);
        gemm_bias_kernel<<<gD, 256, 0, stream>>>(lnb, wSAK + oDD, P[SAKB] + oD,
                                                 nullptr, nullptr, kb, RQ, ADIM, DD, 0);
        gemm_bias_kernel<<<gD, 256, 0, stream>>>(lnb, wSAV + oDD, P[SAVB] + oD,
                                                 nullptr, nullptr, vb, RQ, ADIM, DD, 0);
        attn_kernel<<<gAttn, 128, 0, stream>>>(qb, kb, vb, ctx, x_lens,
                                               TT, TT, 1, scale);
        gemm_bias_kernel<<<gD, 256, 0, stream>>>(ctx, wSAO + oDD, P[SAOB] + oD,
                                                 h, h, nullptr, RQ, DD, ADIM, 0);

        // --- cross attention ---
        layernorm_kernel<<<RQ, 256, 0, stream>>>(h, P[LN2W] + oD, P[LN2B] + oD, lnb);
        gemm_bias_kernel<<<gD, 256, 0, stream>>>(lnb, wCAQ + oDD, P[CAQB] + oD,
                                                 nullptr, nullptr, qb, RQ, ADIM, DD, 0);
        gemm_bias_kernel<<<gDM, 256, 0, stream>>>(mbf, wCAK + oDD, P[CAKB] + oD,
                                                  nullptr, nullptr, kb, RM, ADIM, DD, 0);
        gemm_bias_kernel<<<gDM, 256, 0, stream>>>(mbf, wCAV + oDD, P[CAVB] + oD,
                                                  nullptr, nullptr, vb, RM, ADIM, DD, 0);
        attn_kernel<<<gAttn, 128, 0, stream>>>(qb, kb, vb, ctx, mem_lens,
                                               TT, SS, 0, scale);
        gemm_bias_kernel<<<gD, 256, 0, stream>>>(ctx, wCAO + oDD, P[CAOB] + oD,
                                                 h, h, nullptr, RQ, DD, ADIM, 0);

        // --- FFN (Swish) ---
        layernorm_kernel<<<RQ, 256, 0, stream>>>(h, P[LN3W] + oD, P[LN3B] + oD, lnb);
        gemm_bias_kernel<<<gFF1, 256, 0, stream>>>(lnb, wFF1 + oFD, P[FF1B] + oF,
                                                   nullptr, nullptr, ffb, RQ, FFD, DD, 1);
        gemm_bias_kernel<<<gD, 256, 0, stream>>>(ffb, wFF2 + oFD, P[FF2B] + oD,
                                                 h, h, nullptr, RQ, DD, FFD, 0);
    }

    // ---- vocab projection:  [4096, 512] x [8000, 512]^T -> f32 logits ----
    cvt(h, hbf, (size_t)RQ * DD);
    const dim3 gOut((VV + 127) / 128, RQ / 128);   // (63, 32)
    gemm_bias_kernel<<<gOut, 256, 0, stream>>>(hbf, wOUT, out_b, nullptr,
                                               (float*)d_out, nullptr, RQ, VV, DD, 0);
}
